// TransformerScaledDotProductAttention_73959336837600
// MI455X (gfx1250) — matrix operand order repaired and checked by reading, not compiled, not run on hardware
//
#include <hip/hip_runtime.h>
#include <math.h>

// Problem constants (B,H,L,D) = (4,16,2048,64), fp32 in/out.
#define ATT_B 4
#define ATT_H 16
#define ATT_L 2048
#define ATT_D 64
#define ATT_N (ATT_B * ATT_H * ATT_L * ATT_D)  // elements per tensor

typedef __bf16 bf16_t;
typedef __attribute__((ext_vector_type(16))) __bf16 bf16x16;
typedef __attribute__((ext_vector_type(8)))  __bf16 bf16x8;
typedef __attribute__((ext_vector_type(8)))  float  f32x8;

#define F32X8_ZERO ((f32x8){0.f, 0.f, 0.f, 0.f, 0.f, 0.f, 0.f, 0.f})

static __device__ inline bf16x16 cat8(bf16x8 a, bf16x8 b) {
  return __builtin_shufflevector(a, b, 0, 1, 2, 3, 4, 5, 6, 7,
                                        8, 9, 10, 11, 12, 13, 14, 15);
}

// A-fragment (16x32 bf16): lane-half picks K-offset 0/16; element halves pick
// +0/+8 -> two contiguous bf16x8 runs per lane (data already bf16).
static __device__ inline bf16x16 load_afrag(const bf16_t* row, int chunk, int half) {
  const bf16_t* p = row + chunk * 32 + half * 16;
  bf16x8 lo = *reinterpret_cast<const bf16x8*>(p);
  bf16x8 hi = *reinterpret_cast<const bf16x8*>(p + 8);
  return cat8(lo, hi);
}

// B-fragment (32x16 bf16): lane holds column N=lane%16, contiguous 16 K values.
static __device__ inline bf16x16 load_bfrag(const bf16_t* row, int chunk, int half) {
  return *reinterpret_cast<const bf16x16*>(row + chunk * 32 + half * 16);
}

#define WMMA_BF16(a, b, c) \
  __builtin_amdgcn_wmma_f32_16x16x32_bf16(false, (a), false, (b), (short)0, (c), false, false)

// ---------------------------------------------------------------------------
// Kernel 0: f32 -> bf16 conversion (optionally scaled); 8 elems per iter.
// ---------------------------------------------------------------------------
__global__ __launch_bounds__(256) void cvt_bf16_k(const float* __restrict__ src,
                                                  bf16_t* __restrict__ dst,
                                                  int n8, float scale) {
  int idx = blockIdx.x * blockDim.x + threadIdx.x;
  int stride = gridDim.x * blockDim.x;
  for (int i = idx; i < n8; i += stride) {
    const float4* p = reinterpret_cast<const float4*>(src) + (size_t)i * 2;
    float4 a = p[0], b = p[1];
    bf16x8 o;
    o[0] = (bf16_t)(a.x * scale); o[1] = (bf16_t)(a.y * scale);
    o[2] = (bf16_t)(a.z * scale); o[3] = (bf16_t)(a.w * scale);
    o[4] = (bf16_t)(b.x * scale); o[5] = (bf16_t)(b.y * scale);
    o[6] = (bf16_t)(b.z * scale); o[7] = (bf16_t)(b.w * scale);
    reinterpret_cast<bf16x8*>(dst)[i] = o;
  }
}

// ---------------------------------------------------------------------------
// Kernel 1: softmax coefficient c[q] = m[q] + ln(Z[q]) for rows of
// S = (Q*scale) K^T. One wave owns 32 queries (two 16-row S accumulators
// sharing each K B-fragment -> 4 WMMAs per K fetch). Two sweeps (max, then
// sum-exp) with per-lane partials; cross-lane xor-reduction only at the end.
// K fragments are double-buffered so loads overlap WMMA.
// ---------------------------------------------------------------------------
__global__ __launch_bounds__(256) void attn_stats_k(const bf16_t* __restrict__ qb,
                                                    const bf16_t* __restrict__ kb,
                                                    float* __restrict__ cbuf) {
  const int lane = threadIdx.x & 31;
  const int half = lane >> 4;
  const int lm   = lane & 15;
  const int wave = blockIdx.x * (blockDim.x >> 5) + (threadIdx.x >> 5);
  const int QT   = ATT_L / 32;
  const int qt   = wave % QT;   // 32-query tile
  const int bh   = wave / QT;

  const bf16_t* Qh = qb + (size_t)bh * ATT_L * ATT_D;
  const bf16_t* Kh = kb + (size_t)bh * ATT_L * ATT_D;

  bf16x16 aQ[2][2];
#pragma unroll
  for (int s = 0; s < 2; ++s) {
    const bf16_t* qrow = Qh + (size_t)(qt * 32 + s * 16 + lm) * ATT_D;
    aQ[s][0] = load_afrag(qrow, 0, half);
    aQ[s][1] = load_afrag(qrow, 1, half);
  }

  float mr[2][8], zr[2][8];
#pragma unroll
  for (int s = 0; s < 2; ++s)
#pragma unroll
    for (int i = 0; i < 8; ++i) { mr[s][i] = -INFINITY; zr[s][i] = 0.0f; }

  // ---- Sweep A: per-lane running row max ----
  {
    const bf16_t* krow = Kh + (size_t)lm * ATT_D;
    bf16x16 b0 = load_bfrag(krow, 0, half);
    bf16x16 b1 = load_bfrag(krow, 1, half);
    for (int k0 = 0; k0 < ATT_L; k0 += 16) {
      bf16x16 nb0 = b0, nb1 = b1;
      if (k0 + 16 < ATT_L) {  // prefetch next tile's fragments into registers
        const bf16_t* nk = Kh + (size_t)(k0 + 16 + lm) * ATT_D;
        nb0 = load_bfrag(nk, 0, half);
        nb1 = load_bfrag(nk, 1, half);
      }
      f32x8 a0 = F32X8_ZERO, a1 = F32X8_ZERO;
      a0 = WMMA_BF16(aQ[0][0], b0, a0);
      a1 = WMMA_BF16(aQ[1][0], b0, a1);
      a0 = WMMA_BF16(aQ[0][1], b1, a0);
      a1 = WMMA_BF16(aQ[1][1], b1, a1);
#pragma unroll
      for (int i = 0; i < 8; ++i) {
        mr[0][i] = fmaxf(mr[0][i], a0[i]);
        mr[1][i] = fmaxf(mr[1][i], a1[i]);
      }
      b0 = nb0;
      b1 = nb1;
    }
  }
#pragma unroll
  for (int s = 0; s < 2; ++s)
#pragma unroll
    for (int i = 0; i < 8; ++i) {
      mr[s][i] = fmaxf(mr[s][i], __shfl_xor(mr[s][i], 1));
      mr[s][i] = fmaxf(mr[s][i], __shfl_xor(mr[s][i], 2));
      mr[s][i] = fmaxf(mr[s][i], __shfl_xor(mr[s][i], 4));
      mr[s][i] = fmaxf(mr[s][i], __shfl_xor(mr[s][i], 8));
    }

  // ---- Sweep B: per-lane partial sum of exp(s - m) ----
  {
    const bf16_t* krow = Kh + (size_t)lm * ATT_D;
    bf16x16 b0 = load_bfrag(krow, 0, half);
    bf16x16 b1 = load_bfrag(krow, 1, half);
    for (int k0 = 0; k0 < ATT_L; k0 += 16) {
      bf16x16 nb0 = b0, nb1 = b1;
      if (k0 + 16 < ATT_L) {
        const bf16_t* nk = Kh + (size_t)(k0 + 16 + lm) * ATT_D;
        nb0 = load_bfrag(nk, 0, half);
        nb1 = load_bfrag(nk, 1, half);
      }
      f32x8 a0 = F32X8_ZERO, a1 = F32X8_ZERO;
      a0 = WMMA_BF16(aQ[0][0], b0, a0);
      a1 = WMMA_BF16(aQ[1][0], b0, a1);
      a0 = WMMA_BF16(aQ[0][1], b1, a0);
      a1 = WMMA_BF16(aQ[1][1], b1, a1);
#pragma unroll
      for (int i = 0; i < 8; ++i) {
        zr[0][i] += __expf(a0[i] - mr[0][i]);
        zr[1][i] += __expf(a1[i] - mr[1][i]);
      }
      b0 = nb0;
      b1 = nb1;
    }
  }
#pragma unroll
  for (int s = 0; s < 2; ++s)
#pragma unroll
    for (int i = 0; i < 8; ++i) {
      zr[s][i] += __shfl_xor(zr[s][i], 1);
      zr[s][i] += __shfl_xor(zr[s][i], 2);
      zr[s][i] += __shfl_xor(zr[s][i], 4);
      zr[s][i] += __shfl_xor(zr[s][i], 8);
    }

  if (lm == 0) {  // lanes 0 and 16 each own 8 rows per sub-tile
#pragma unroll
    for (int s = 0; s < 2; ++s)
#pragma unroll
      for (int i = 0; i < 8; ++i) {
        int q = qt * 32 + s * 16 + half * 8 + i;
        cbuf[(size_t)bh * ATT_L + q] = mr[s][i] + __logf(zr[s][i]);
      }
  }
}

// ---------------------------------------------------------------------------
// Kernel 2: out[d,k] = sum_q V[d,q] * exp(S[q,k] - c[q]).
// One wave owns 32 output columns (two 16-wide k tiles): Q A-fragments and
// V fragments are fetched once and reused by both k tiles -> 16 WMMAs per
// 32-query step. P staged through LDS in [k][q] order to become the 32x16
// B-fragment of the second matmul.
// ---------------------------------------------------------------------------
__global__ __launch_bounds__(256) void attn_out_k(const bf16_t* __restrict__ qb,
                                                  const bf16_t* __restrict__ kb,
                                                  const bf16_t* __restrict__ vb,
                                                  const float* __restrict__ cbuf,
                                                  float* __restrict__ out) {
  __shared__ __align__(16) unsigned short lds[8][2][32 * 16];  // 16 KB

  const int lane = threadIdx.x & 31;
  const int half = lane >> 4;
  const int lm   = lane & 15;
  const int w    = threadIdx.x >> 5;
  const int wave = blockIdx.x * (blockDim.x >> 5) + w;
  const int KT   = ATT_L / 32;
  const int kt   = wave % KT;   // 32-column tile
  const int bh   = wave / KT;

  const bf16_t* Qh = qb + (size_t)bh * ATT_L * ATT_D;
  const bf16_t* Kh = kb + (size_t)bh * ATT_L * ATT_D;
  const bf16_t* Vh = vb + (size_t)bh * ATT_D * ATT_L;
  const float*  ch = cbuf + (size_t)bh * ATT_L;

  // K^T B-fragments for both k tiles: invariant over q, hoisted.
  bf16x16 bK[2][2];
#pragma unroll
  for (int t = 0; t < 2; ++t) {
    const bf16_t* krow = Kh + (size_t)(kt * 32 + t * 16 + lm) * ATT_D;
    bK[t][0] = load_bfrag(krow, 0, half);
    bK[t][1] = load_bfrag(krow, 1, half);
  }

  f32x8 oacc[2][4];
#pragma unroll
  for (int t = 0; t < 2; ++t)
#pragma unroll
    for (int db = 0; db < 4; ++db) oacc[t][db] = F32X8_ZERO;

  for (int q0 = 0; q0 < ATT_L; q0 += 32) {
#pragma unroll
    for (int sub = 0; sub < 2; ++sub) {
      const int qbase = q0 + sub * 16;
      const bf16_t* qrow = Qh + (size_t)(qbase + lm) * ATT_D;
      bf16x16 a0 = load_afrag(qrow, 0, half);
      bf16x16 a1 = load_afrag(qrow, 1, half);

      f32x8 s0 = F32X8_ZERO, s1 = F32X8_ZERO;
      s0 = WMMA_BF16(a0, bK[0][0], s0);
      s1 = WMMA_BF16(a0, bK[1][0], s1);
      s0 = WMMA_BF16(a1, bK[0][1], s0);
      s1 = WMMA_BF16(a1, bK[1][1], s1);

      // c[q] for this lane's 8 rows (shared by both k tiles).
      const float* cp = ch + qbase + half * 8;
      float4 c0 = *reinterpret_cast<const float4*>(cp);
      float4 c1 = *reinterpret_cast<const float4*>(cp + 4);
      float cr[8] = {c0.x, c0.y, c0.z, c0.w, c1.x, c1.y, c1.z, c1.w};

      bf16x8 pv0, pv1;
#pragma unroll
      for (int i = 0; i < 8; ++i) {
        pv0[i] = (bf16_t)__expf(s0[i] - cr[i]);
        pv1[i] = (bf16_t)__expf(s1[i] - cr[i]);
      }
      // [k][q] staging stores: 8 consecutive bf16 -> ds_store_b128 each.
      const int off = lm * 32 + sub * 16 + half * 8;
      *reinterpret_cast<bf16x8*>(&lds[w][0][off]) = pv0;
      *reinterpret_cast<bf16x8*>(&lds[w][1][off]) = pv1;
    }

    asm volatile("s_wait_dscnt 0" ::: "memory");  // same-wave LDS RAW

    // Read back as 32x16 B-fragments: lane = column N=lm, lane-half picks
    // q-range 0-15 / 16-31 -> 16 consecutive bf16 in [k][q] order.
    bf16x16 pf0 = *reinterpret_cast<const bf16x16*>(&lds[w][0][lm * 32 + half * 16]);
    bf16x16 pf1 = *reinterpret_cast<const bf16x16*>(&lds[w][1][lm * 32 + half * 16]);

    // O(64x32) += V_blk(16x32) * P(32x16) per k tile; V fetched once.
#pragma unroll
    for (int db = 0; db < 4; ++db) {
      const bf16_t* vrow = Vh + (size_t)(db * 16 + lm) * ATT_L + q0 + half * 16;
      bf16x8 lo = *reinterpret_cast<const bf16x8*>(vrow);
      bf16x8 hi = *reinterpret_cast<const bf16x8*>(vrow + 8);
      bf16x16 av = cat8(lo, hi);
      oacc[0][db] = WMMA_BF16(av, pf0, oacc[0][db]);
      oacc[1][db] = WMMA_BF16(av, pf1, oacc[1][db]);
    }
  }

#pragma unroll
  for (int t = 0; t < 2; ++t)
#pragma unroll
    for (int db = 0; db < 4; ++db)
#pragma unroll
      for (int i = 0; i < 8; ++i) {
        const int d = db * 16 + half * 8 + i;
        out[((size_t)bh * ATT_D + d) * ATT_L + kt * 32 + t * 16 + lm] = oacc[t][db][i];
      }
}

// ---------------------------------------------------------------------------
// Workspace layout: bf16 Q (scaled) | bf16 K | bf16 V | f32 c[q]
//   3 * 16.78 MB + 0.52 MB ~= 50.9 MB of d_ws.
// ---------------------------------------------------------------------------
extern "C" void kernel_launch(void* const* d_in, const int* in_sizes, int n_in,
                              void* d_out, int out_size, void* d_ws, size_t ws_size,
                              hipStream_t stream) {
  (void)in_sizes; (void)n_in; (void)out_size; (void)ws_size;
  const float* Q = (const float*)d_in[0];  // [B,H,L,D]
  const float* K = (const float*)d_in[1];  // [B,H,L,D]
  const float* V = (const float*)d_in[2];  // [B,H,D,L]
  float* out = (float*)d_out;              // [B,H,D,L]

  bf16_t* qbuf = (bf16_t*)d_ws;
  bf16_t* kbuf = qbuf + (size_t)ATT_N;
  bf16_t* vbuf = kbuf + (size_t)ATT_N;
  float*  cb   = (float*)(vbuf + (size_t)ATT_N);

  const int n8 = ATT_N / 8;
  cvt_bf16_k<<<2048, 256, 0, stream>>>(Q, qbuf, n8, 0.125f);  // fold 1/sqrt(D)
  cvt_bf16_k<<<2048, 256, 0, stream>>>(K, kbuf, n8, 1.0f);
  cvt_bf16_k<<<2048, 256, 0, stream>>>(V, vbuf, n8, 1.0f);

  const int waves = ATT_B * ATT_H * (ATT_L / 32);  // 4096 wave-tiles
  dim3 block(256);
  dim3 grid(waves / 8);                            // 512 blocks
  attn_stats_k<<<grid, block, 0, stream>>>(qbuf, kbuf, cb);
  attn_out_k<<<grid, block, 0, stream>>>(qbuf, kbuf, vbuf, cb, out);
}
